// MessagePassingGNN_82257213653679
// MI455X (gfx1250) — matrix operand order repaired and checked by reading, not compiled, not run on hardware
//
#include <hip/hip_runtime.h>
#include <hip/hip_bf16.h>
#include <math.h>

// ---------------------------------------------------------------------------
// MessagePassingGNN for MI455X (gfx1250, wave32, WMMA).
//
// Roofline: ~157 GFLOP f16 GEMM vs ~3.7 GB traffic -> bandwidth bound
// (~160us at 23.3 TB/s). f16 activations feed v_wmma_f32_16x16x32_f16;
// fp32 accumulate + tanh epilogue; segment_sum via global_atomic_add_f32.
//
// Round 3: round 2's LDS-staged weights were defeated by LICM (the ds_loads
// were hoisted, re-inflating B to 128 VGPRs and spilling to scratch inside
// the WMMA loop). B fragments are now loaded with asm volatile ds_load_b128,
// which LICM cannot hoist -> ~70 live VGPRs, no spills, high occupancy.
// ---------------------------------------------------------------------------

#define NNODES 50000
#define NEDGES 1600000
#define DDIM   64
#define NLAY   3

typedef _Float16 half16 __attribute__((ext_vector_type(16)));
typedef _Float16 half8  __attribute__((ext_vector_type(8)));
typedef float    float8 __attribute__((ext_vector_type(8)));
typedef float    float4v __attribute__((ext_vector_type(4)));

__device__ __forceinline__ half16 cat16(half8 lo, half8 hi) {
    half16 r;
#pragma unroll
    for (int i = 0; i < 8; ++i) { r[i] = lo[i]; r[i + 8] = hi[i]; }
    return r;
}

// Two 16-byte ds_load_b128 at byte offsets 0 and 32, opaque to LICM (asm volatile).
// LDS byte offset = low 32 bits of the generic pointer (ISA LDS aperture).
__device__ __forceinline__ half16 lds_load_bfrag(const _Float16* p) {
    unsigned off = (unsigned)(unsigned long long)p;
    half8 lo, hi;
    asm volatile("ds_load_b128 %0, %2\n"
                 "ds_load_b128 %1, %2 offset:32\n"
                 "s_wait_dscnt 0x0"
                 : "=v"(lo), "=v"(hi)
                 : "v"(off));
    return cat16(lo, hi);
}

// ---- one-time per-launch conversions ---------------------------------------

// W_msg/W_upd: fp32 [L][128][64] -> f16 transposed [L][64][128] (N-major so
// each lane's B fragment is two 16-byte runs of one W^T row).
__global__ void wcvt_kernel(const float* __restrict__ Wmsg,
                            const float* __restrict__ Wupd,
                            _Float16* __restrict__ WtMsg,
                            _Float16* __restrict__ WtUpd) {
    const int per = NLAY * 128 * DDIM;
    int i = blockIdx.x * blockDim.x + threadIdx.x;
    if (i < per) {
        int l = i / (128 * DDIM), r = i % (128 * DDIM), k = r / DDIM, n = r % DDIM;
        WtMsg[l * DDIM * 128 + n * 128 + k] = (_Float16)Wmsg[i];
    } else if (i < 2 * per) {
        int j = i - per;
        int l = j / (128 * DDIM), r = j % (128 * DDIM), k = r / DDIM, n = r % DDIM;
        WtUpd[l * DDIM * 128 + n * 128 + k] = (_Float16)Wupd[j];
    }
}

__global__ void cvt_h_kernel(const float* __restrict__ nf, _Float16* __restrict__ h) {
    int i = blockIdx.x * blockDim.x + threadIdx.x;
    if (i < NNODES * DDIM) h[i] = (_Float16)nf[i];
}

// e = tanh(edge_feature @ W_emb + b_emb), stored f16. IN_DIM=2 so plain VALU.
__global__ void edge_embed_kernel(const float* __restrict__ ef,
                                  const float* __restrict__ Wemb,
                                  const float* __restrict__ bemb,
                                  _Float16* __restrict__ e) {
    long i = (long)blockIdx.x * blockDim.x + threadIdx.x;
    if (i >= (long)NEDGES * DDIM) return;
    int edge = (int)(i >> 6), c = (int)(i & 63);
    float v = ef[2 * edge] * Wemb[c] + ef[2 * edge + 1] * Wemb[DDIM + c] + bemb[c];
    e[i] = (_Float16)tanhf(v);
}

__global__ void zero_kernel(float* __restrict__ p, int n) {
    int i = blockIdx.x * blockDim.x + threadIdx.x;
    if (i < n) p[i] = 0.0f;
}

// ---- layer kernels: 16-row x 64-col tiles via v_wmma_f32_16x16x32_f16 ------
//
// A fragment (16x32 f16): lane&15 = row; K chunks [base..base+7],[base+16..+23]
//   with base = kk*32 + (lane>=16 ? 8 : 0)  -> two b128 global loads.
// B fragment (32x16 f16): lane&15 = col N; K = kk*32 + (lane>=16?8:0) + t + (t>=8?8:0)
//   -> two 16-byte runs of W^T[n][*], 32 B apart     -> two ds_load_b128 per WMMA.
// C/D (16x16 f32): row = v + (lane>=16?8:0), col = lane&15.

__device__ __forceinline__ void load_weights_to_lds(_Float16* sW,
                                                    const _Float16* __restrict__ Wt,
                                                    int tid, int nthreads) {
    // 64*128 halves = 16KB = 1024 uint4 chunks
    const uint4* src = (const uint4*)Wt;
    uint4* dst = (uint4*)sW;
    for (int i = tid; i < (DDIM * 128 * 2) / 16; i += nthreads) dst[i] = src[i];
}

__global__ void __launch_bounds__(256)
msg_kernel(const _Float16* __restrict__ h,
           _Float16* __restrict__ e,           // read + in-place write (e = m)
           const int* __restrict__ src,
           const int* __restrict__ dst,
           const _Float16* __restrict__ Wt,    // [64][128] f16, this layer
           const float* __restrict__ bias,     // [64]
           float* __restrict__ aggr,           // [N][64] fp32, pre-zeroed
           int ntiles) {
    __shared__ __align__(16) _Float16 sW[DDIM * 128];

    load_weights_to_lds(sW, Wt, threadIdx.x, blockDim.x);
    __syncthreads();

    const int lane = threadIdx.x & 31;
    const int wv = threadIdx.x >> 5;
    const int wavesPerBlk = blockDim.x >> 5;
    const int gwave = blockIdx.x * wavesPerBlk + wv;
    const int nwaves = gridDim.x * wavesPerBlk;
    const int ln15 = lane & 15;
    const int hi16 = (lane >> 4) & 1;

    float bval[4];
#pragma unroll
    for (int nn = 0; nn < 4; ++nn) bval[nn] = bias[nn * 16 + ln15];

    // per-lane base of this lane's B-fragment region in LDS (halves)
    const _Float16* sWlane = sW + ln15 * 128 + hi16 * 8;
    const int rowOutBase = hi16 * 8;

    for (int tile = gwave; tile < ntiles; tile += nwaves) {
        const int aEdge = tile * 16 + ln15;
        const int sNode = src[aEdge];
        const _Float16* hrow = h + (long)sNode * DDIM;
        const _Float16* erow = e + (long)aEdge * DDIM;

        float8 acc[4];
#pragma unroll
        for (int nn = 0; nn < 4; ++nn)
#pragma unroll
            for (int i = 0; i < 8; ++i) acc[nn][i] = bval[nn];

#pragma unroll
        for (int kk = 0; kk < 4; ++kk) {
            const int K0 = kk * 32 + hi16 * 8;
            const _Float16* p = (kk < 2) ? (hrow + K0) : (erow + (K0 - 64));
            half16 Af = cat16(*(const half8*)p, *(const half8*)(p + 16));
#pragma unroll
            for (int nn = 0; nn < 4; ++nn) {
                half16 Bf = lds_load_bfrag(sWlane + nn * 16 * 128 + kk * 32);
                acc[nn] = __builtin_amdgcn_wmma_f32_16x16x32_f16(
                    false, Af, false, Bf, (short)0, acc[nn], false, false);
            }
        }

        // epilogue: tanh, store m into e (in-place), atomic scatter into aggr[dst]
#pragma unroll
        for (int v = 0; v < 8; ++v) {
            const int oEdge = tile * 16 + rowOutBase + v;
            const int dNode = dst[oEdge];
#pragma unroll
            for (int nn = 0; nn < 4; ++nn) {
                const int col = nn * 16 + ln15;
                const float val = tanhf(acc[nn][v]);
                e[(long)oEdge * DDIM + col] = (_Float16)val;
                atomicAdd(aggr + (long)dNode * DDIM + col, val);
            }
        }
    }
}

__global__ void __launch_bounds__(256)
upd_kernel(const _Float16* __restrict__ hin,
           const float* __restrict__ aggr,
           _Float16* __restrict__ hout,
           const _Float16* __restrict__ Wt,   // [64][128]
           const float* __restrict__ bias,    // [64]
           int ntiles) {
    __shared__ __align__(16) _Float16 sW[DDIM * 128];

    load_weights_to_lds(sW, Wt, threadIdx.x, blockDim.x);
    __syncthreads();

    const int lane = threadIdx.x & 31;
    const int wv = threadIdx.x >> 5;
    const int wavesPerBlk = blockDim.x >> 5;
    const int gwave = blockIdx.x * wavesPerBlk + wv;
    const int nwaves = gridDim.x * wavesPerBlk;
    const int ln15 = lane & 15;
    const int hi16 = (lane >> 4) & 1;

    float bval[4];
#pragma unroll
    for (int nn = 0; nn < 4; ++nn) bval[nn] = bias[nn * 16 + ln15];

    const _Float16* sWlane = sW + ln15 * 128 + hi16 * 8;
    const int rowOutBase = hi16 * 8;

    for (int tile = gwave; tile < ntiles; tile += nwaves) {
        const int node = tile * 16 + ln15;
        const _Float16* hrow = hin + (long)node * DDIM;
        const float* arow = aggr + (long)node * DDIM;

        float8 acc[4];
#pragma unroll
        for (int nn = 0; nn < 4; ++nn)
#pragma unroll
            for (int i = 0; i < 8; ++i) acc[nn][i] = bval[nn];

#pragma unroll
        for (int kk = 0; kk < 4; ++kk) {
            const int K0 = kk * 32 + hi16 * 8;
            half16 Af;
            if (kk < 2) {
                const _Float16* p = hrow + K0;
                Af = cat16(*(const half8*)p, *(const half8*)(p + 16));
            } else {
                const int c0 = K0 - 64;
                float4v f0 = *(const float4v*)(arow + c0);
                float4v f1 = *(const float4v*)(arow + c0 + 4);
                float4v f2 = *(const float4v*)(arow + c0 + 16);
                float4v f3 = *(const float4v*)(arow + c0 + 20);
#pragma unroll
                for (int i = 0; i < 4; ++i) {
                    Af[i]      = (_Float16)f0[i];
                    Af[4 + i]  = (_Float16)f1[i];
                    Af[8 + i]  = (_Float16)f2[i];
                    Af[12 + i] = (_Float16)f3[i];
                }
            }
#pragma unroll
            for (int nn = 0; nn < 4; ++nn) {
                half16 Bf = lds_load_bfrag(sWlane + nn * 16 * 128 + kk * 32);
                acc[nn] = __builtin_amdgcn_wmma_f32_16x16x32_f16(
                    false, Af, false, Bf, (short)0, acc[nn], false, false);
            }
        }

#pragma unroll
        for (int v = 0; v < 8; ++v) {
            const int oNode = tile * 16 + rowOutBase + v;
#pragma unroll
            for (int nn = 0; nn < 4; ++nn) {
                const int col = nn * 16 + ln15;
                hout[(long)oNode * DDIM + col] = (_Float16)tanhf(acc[nn][v]);
            }
        }
    }
}

// decoder: out = tanh(h @ W_d1 + b_d1) @ W_d2 + b_d2  (tiny: VALU per node)
__global__ void dec_kernel(const _Float16* __restrict__ h,
                           const float* __restrict__ W1, const float* __restrict__ b1,
                           const float* __restrict__ W2, const float* __restrict__ b2,
                           float* __restrict__ out) {
    int n = blockIdx.x * blockDim.x + threadIdx.x;
    if (n >= NNODES) return;
    float hv[DDIM];
#pragma unroll
    for (int k = 0; k < DDIM; ++k) hv[k] = (float)h[(long)n * DDIM + k];
    float o = b2[0];
    for (int j = 0; j < DDIM / 2; ++j) {
        float a = b1[j];
#pragma unroll
        for (int k = 0; k < DDIM; ++k) a += hv[k] * W1[k * (DDIM / 2) + j];
        o += tanhf(a) * W2[j];
    }
    out[n] = o;
}

// ---------------------------------------------------------------------------

extern "C" void kernel_launch(void* const* d_in, const int* in_sizes, int n_in,
                              void* d_out, int out_size, void* d_ws, size_t ws_size,
                              hipStream_t stream) {
    const float* node_feature = (const float*)d_in[0];
    const int*   edge_index   = (const int*)d_in[1];   // [2][E]
    const float* edge_feature = (const float*)d_in[2];
    /* d_in[3] = batch (unused) */
    const float* W_emb = (const float*)d_in[4];
    const float* b_emb = (const float*)d_in[5];
    const float* W_msg = (const float*)d_in[6];
    const float* b_msg = (const float*)d_in[7];
    const float* W_upd = (const float*)d_in[8];
    const float* b_upd = (const float*)d_in[9];
    const float* W_d1  = (const float*)d_in[10];
    const float* b_d1  = (const float*)d_in[11];
    const float* W_d2  = (const float*)d_in[12];
    const float* b_d2  = (const float*)d_in[13];

    // scratch carve-out (256B aligned): e (f16), h0/h1 (f16), aggr (f32), Wt's
    char* ws = (char*)d_ws;
    size_t off = 0;
    auto carve = [&](size_t bytes) -> char* {
        char* p = ws + off;
        off = (off + bytes + 255) & ~(size_t)255;
        return p;
    };
    _Float16* e_buf = (_Float16*)carve((size_t)NEDGES * DDIM * 2);
    _Float16* h0    = (_Float16*)carve((size_t)NNODES * DDIM * 2);
    _Float16* h1    = (_Float16*)carve((size_t)NNODES * DDIM * 2);
    float*    aggr  = (float*)   carve((size_t)NNODES * DDIM * 4);
    _Float16* WtMsg = (_Float16*)carve((size_t)NLAY * DDIM * 128 * 2);
    _Float16* WtUpd = (_Float16*)carve((size_t)NLAY * DDIM * 128 * 2);
    (void)ws_size; (void)in_sizes; (void)n_in; (void)out_size;

    const int* src = edge_index;
    const int* dst = edge_index + NEDGES;

    wcvt_kernel<<<(2 * NLAY * 128 * DDIM + 255) / 256, 256, 0, stream>>>(
        W_msg, W_upd, WtMsg, WtUpd);
    cvt_h_kernel<<<(NNODES * DDIM + 255) / 256, 256, 0, stream>>>(node_feature, h0);
    edge_embed_kernel<<<(int)(((long)NEDGES * DDIM + 255) / 256), 256, 0, stream>>>(
        edge_feature, W_emb, b_emb, e_buf);

    _Float16* hc = h0;
    _Float16* hn = h1;
    for (int l = 0; l < NLAY; ++l) {
        zero_kernel<<<(NNODES * DDIM + 255) / 256, 256, 0, stream>>>(aggr, NNODES * DDIM);
        msg_kernel<<<1250, 256, 0, stream>>>(hc, e_buf, src, dst,
                                             WtMsg + (size_t)l * DDIM * 128,
                                             b_msg + (size_t)l * DDIM,
                                             aggr, NEDGES / 16);
        upd_kernel<<<200, 256, 0, stream>>>(hc, aggr, hn,
                                            WtUpd + (size_t)l * DDIM * 128,
                                            b_upd + (size_t)l * DDIM,
                                            NNODES / 16);
        _Float16* t = hc; hc = hn; hn = t;
    }
    dec_kernel<<<(NNODES + 255) / 256, 256, 0, stream>>>(hc, W_d1, b_d1, W_d2, b_d2,
                                                         (float*)d_out);
}
